// LSTMReg_73100343378275
// MI455X (gfx1250) — compile-verified, matrix operand order checked by reading, not run on hardware
//
#include <hip/hip_runtime.h>

// ---------------------------------------------------------------------------
// Fused 2-layer LSTM (B=512, T=1024, F=H=64) + final FC, for gfx1250 (CDNA5).
// 32 workgroups x 128 threads (4 wave32 waves). Each workgroup owns a 16-row
// batch tile and runs the whole serial recurrence for both layers with
// v_wmma_f32_16x16x32_f16. ALL weight B-operands live in VGPRs (32 x v16h =
// 256 VGPRs/lane, loaded once), cell state lives in VGPRs, and h round-trips
// through a tiny LDS tile (the D-layout -> A-layout transpose). x is
// pre-tiled to f16 row-major 16x64 tiles per (bblk,t) and software-pipelined.
// Per step/layer: A = [x_t | h_prev] (16x128), B rows = [W_ih|W_hh] (256x128),
// wave w owns hidden cols [16w,16w+16) of ALL four gates so gate math is pure
// per-lane VALU/TRANS (no shuffles), co-executing with the XDL pipe.
// ---------------------------------------------------------------------------

typedef _Float16 v16h __attribute__((ext_vector_type(16)));
typedef _Float16 v8h  __attribute__((ext_vector_type(8)));
typedef float    v8f  __attribute__((ext_vector_type(8)));

#define BB 512
#define TT 1024
#define FF 64
#define HH 64

#if __has_builtin(__builtin_amdgcn_tanhf)
__device__ __forceinline__ float fast_tanh(float x) { return __builtin_amdgcn_tanhf(x); }
#else
__device__ __forceinline__ float fast_tanh(float x) { return tanhf(x); }
#endif

// sigmoid(x) = 0.5*tanh(0.5x)+0.5  (one TRANS op instead of exp+rcp)
__device__ __forceinline__ float sigm(float x) {
    return __builtin_fmaf(0.5f, fast_tanh(0.5f * x), 0.5f);
}

// Build a 16-half A/B operand from two contiguous 16B loads.
// ISA 16-bit operand layout: lanes 0-15 hold K = [k0..k0+7, k0+16..k0+23],
// lanes 16-31 hold [k0+8..k0+15, k0+24..k0+31]; caller pre-offsets p by
// (lane>=16 ? 8 : 0) halves.
__device__ __forceinline__ v16h mk_ab(const _Float16* p) {
    v8h lo = *(const v8h*)(p);
    v8h hi = *(const v8h*)(p + 16);
    v16h r;
#pragma unroll
    for (int i = 0; i < 8; ++i) { r[i] = lo[i]; r[i + 8] = hi[i]; }
    return r;
}

__device__ __forceinline__ v8f splat8(float x) {
    v8f r;
#pragma unroll
    for (int i = 0; i < 8; ++i) r[i] = x;
    return r;
}

// ---------------------------------------------------------------------------
// Prep: x [B,T,64] f32 -> per (bblk,t) contiguous 16x64 f16 row-major tile.
// ---------------------------------------------------------------------------
__global__ void prep_x_kernel(const float* __restrict__ x,
                              _Float16* __restrict__ xa, int total) {
    int idx = blockIdx.x * blockDim.x + threadIdx.x;
    if (idx >= total) return;
    int k = idx & 63;
    int t = (idx >> 6) & (TT - 1);
    int b = idx >> 16;                       // / (T*F)
    size_t dst = ((size_t)(b >> 4) * TT + t) * 1024 + (size_t)(b & 15) * 64 + k;
    xa[dst] = (_Float16)x[idx];
}

// wcat[l][n][k] f16: k<64 -> w_ih[l][n][k], else w_hh[l][n][k-64].
__global__ void prep_w_kernel(const float* __restrict__ wih0,
                              const float* __restrict__ whh0,
                              const float* __restrict__ wih1,
                              const float* __restrict__ whh1,
                              _Float16* __restrict__ wcat) {
    int idx = blockIdx.x * blockDim.x + threadIdx.x;   // < 2*256*128
    int l   = idx >> 15;
    int rem = idx & 32767;
    int n   = rem >> 7;
    int k   = rem & 127;
    const float* wih = l ? wih1 : wih0;
    const float* whh = l ? whh1 : whh0;
    float v = (k < 64) ? wih[n * 64 + k] : whh[n * 64 + (k - 64)];
    wcat[idx] = (_Float16)v;
}

// bsum[l][n] = b_ih[l][n] + b_hh[l][n]
__global__ void prep_b_kernel(const float* __restrict__ bih0,
                              const float* __restrict__ bhh0,
                              const float* __restrict__ bih1,
                              const float* __restrict__ bhh1,
                              float* __restrict__ bsum) {
    int idx = blockIdx.x * blockDim.x + threadIdx.x;
    if (idx >= 512) return;
    int l = idx >> 8, n = idx & 255;
    bsum[idx] = l ? (bih1[n] + bhh1[n]) : (bih0[n] + bhh0[n]);
}

// ---------------------------------------------------------------------------
// Main fused kernel.
// ---------------------------------------------------------------------------
__global__ __launch_bounds__(128) void lstm_fused_kernel(
    const _Float16* __restrict__ xa,    // [B/16][T][16][64] f16 tiles
    const _Float16* __restrict__ wcat,  // [2][256][128] f16
    const float* __restrict__ bsum,     // [2][256]
    const float* __restrict__ fc_w,     // [64]
    const float* __restrict__ fc_b,     // [1]
    float* __restrict__ out)            // [B]
{
    // h tiles only; rows padded to 72 halves (144B = 9*16B) for bank spread.
    __shared__ _Float16 sH0[16][72];
    __shared__ _Float16 sH1[16][72];

    const int tid  = threadIdx.x;
    const int wv   = tid >> 5;          // wave 0..3
    const int lane = tid & 31;
    const int lrow = lane & 15;         // A-operand row / row-within-N-tile
    const int hi8  = (lane >> 4) << 3;  // 0 or 8: K sub-offset AND D row offset
    const int ncol = (wv << 4) + lrow;  // hidden column 0..63 owned by lane

    for (int idx = tid; idx < 16 * 72; idx += 128) {
        (&sH0[0][0])[idx] = (_Float16)0.0f;
        (&sH1[0][0])[idx] = (_Float16)0.0f;
    }

    // ---- weight B-operands: resident in VGPRs for the whole recurrence ----
    v16h wB[2][4][4];                   // [layer][gate][k-chunk]
#pragma unroll
    for (int l = 0; l < 2; ++l)
#pragma unroll
        for (int g = 0; g < 4; ++g)
#pragma unroll
            for (int c = 0; c < 4; ++c)
                wB[l][g][c] = mk_ab(
                    wcat + (size_t)((l * 256 + g * 64 + ncol) * 128) + c * 32 + hi8);

    // ---- biases, pre-splatted into accumulator-init vectors ----
    v8f bi0[4], bi1[4];
#pragma unroll
    for (int g = 0; g < 4; ++g) {
        bi0[g] = splat8(bsum[g * 64 + ncol]);
        bi1[g] = splat8(bsum[256 + g * 64 + ncol]);
    }

    const v8f z8 = splat8(0.0f);
    v8f c0 = z8, c1 = z8;               // cell states: rows r+hi8, col ncol

    const _Float16* xbase =
        xa + (size_t)blockIdx.x * TT * 1024 + (size_t)lrow * 64 + hi8;

    __syncthreads();                    // sH zero-init visible

    // software pipeline: x A-operands for step 0
    v16h ax0 = mk_ab(xbase);
    v16h ax1 = mk_ab(xbase + 32);

    for (int t = 0; t < TT; ++t) {
        // prefetch next step's x tile (latency hides under both GEMMs)
        const _Float16* xn = xbase + (size_t)((t < TT - 1) ? t + 1 : t) * 1024;
        v16h nax0 = mk_ab(xn);
        v16h nax1 = mk_ab(xn + 32);

        float hn[8];

        // ================= layer 0: z = [x_t | h0] @ W0^T =================
        {
            v8f acc[4] = {bi0[0], bi0[1], bi0[2], bi0[3]};
            v16h ah0 = mk_ab(&sH0[lrow][hi8]);
            v16h ah1 = mk_ab(&sH0[lrow][32 + hi8]);
#pragma unroll
            for (int g = 0; g < 4; ++g) {
                acc[g] = __builtin_amdgcn_wmma_f32_16x16x32_f16(
                    false, ax0, false, wB[0][g][0], (short)0, acc[g], false, false);
                acc[g] = __builtin_amdgcn_wmma_f32_16x16x32_f16(
                    false, ax1, false, wB[0][g][1], (short)0, acc[g], false, false);
                acc[g] = __builtin_amdgcn_wmma_f32_16x16x32_f16(
                    false, ah0, false, wB[0][g][2], (short)0, acc[g], false, false);
                acc[g] = __builtin_amdgcn_wmma_f32_16x16x32_f16(
                    false, ah1, false, wB[0][g][3], (short)0, acc[g], false, false);
            }
#pragma unroll
            for (int r = 0; r < 8; ++r) {
                float ii = sigm(acc[0][r]);
                float ff = sigm(acc[1][r]);
                float gg = fast_tanh(acc[2][r]);
                float oo = sigm(acc[3][r]);
                float cn = __builtin_fmaf(ff, c0[r], ii * gg);
                c0[r] = cn;
                hn[r] = oo * fast_tanh(cn);
            }
        }
        ax0 = nax0; ax1 = nax1;

        __syncthreads();                 // all waves done reading sH0
#pragma unroll
        for (int r = 0; r < 8; ++r) sH0[r + hi8][ncol] = (_Float16)hn[r];
        __syncthreads();                 // new h0 visible

        // ================= layer 1: z = [h0 | h1] @ W1^T ==================
        {
            v8f acc[4] = {bi1[0], bi1[1], bi1[2], bi1[3]};
            v16h a0 = mk_ab(&sH0[lrow][hi8]);
            v16h a1 = mk_ab(&sH0[lrow][32 + hi8]);
            v16h a2 = mk_ab(&sH1[lrow][hi8]);
            v16h a3 = mk_ab(&sH1[lrow][32 + hi8]);
#pragma unroll
            for (int g = 0; g < 4; ++g) {
                acc[g] = __builtin_amdgcn_wmma_f32_16x16x32_f16(
                    false, a0, false, wB[1][g][0], (short)0, acc[g], false, false);
                acc[g] = __builtin_amdgcn_wmma_f32_16x16x32_f16(
                    false, a1, false, wB[1][g][1], (short)0, acc[g], false, false);
                acc[g] = __builtin_amdgcn_wmma_f32_16x16x32_f16(
                    false, a2, false, wB[1][g][2], (short)0, acc[g], false, false);
                acc[g] = __builtin_amdgcn_wmma_f32_16x16x32_f16(
                    false, a3, false, wB[1][g][3], (short)0, acc[g], false, false);
            }
#pragma unroll
            for (int r = 0; r < 8; ++r) {
                float ii = sigm(acc[0][r]);
                float ff = sigm(acc[1][r]);
                float gg = fast_tanh(acc[2][r]);
                float oo = sigm(acc[3][r]);
                float cn = __builtin_fmaf(ff, c1[r], ii * gg);
                c1[r] = cn;
                hn[r] = oo * fast_tanh(cn);
            }
        }
        __syncthreads();                 // all waves done reading sH1
#pragma unroll
        for (int r = 0; r < 8; ++r) sH1[r + hi8][ncol] = (_Float16)hn[r];
        // NOTE: no 4th barrier needed -- the next reader of sH1 (layer 1 of
        // step t+1) sits behind the two barriers at the top of step t+1.
    }

    __syncthreads();

    // ================= final FC on h2[:, T-1] =================
    if (tid < 16) {
        float s = fc_b[0];
#pragma unroll
        for (int n = 0; n < 64; ++n) s += (float)sH1[tid][n] * fc_w[n];
        out[blockIdx.x * 16 + tid] = s;
    }
}

// ---------------------------------------------------------------------------
extern "C" void kernel_launch(void* const* d_in, const int* in_sizes, int n_in,
                              void* d_out, int out_size, void* d_ws, size_t ws_size,
                              hipStream_t stream) {
    const float* x     = (const float*)d_in[0];
    const float* w_ih0 = (const float*)d_in[1];
    const float* w_hh0 = (const float*)d_in[2];
    const float* b_ih0 = (const float*)d_in[3];
    const float* b_hh0 = (const float*)d_in[4];
    const float* w_ih1 = (const float*)d_in[5];
    const float* w_hh1 = (const float*)d_in[6];
    const float* b_ih1 = (const float*)d_in[7];
    const float* b_hh1 = (const float*)d_in[8];
    const float* fc_w  = (const float*)d_in[9];
    const float* fc_b  = (const float*)d_in[10];

    // Workspace: xa (64 MiB) | wcat (128 KiB) | bsum (2 KiB)
    char* ws = (char*)d_ws;
    _Float16* xa   = (_Float16*)ws;
    _Float16* wcat = (_Float16*)(ws + 67108864ull);
    float*    bsmn = (float*)(ws + 67108864ull + 131072ull);

    const int total_x = BB * TT * FF;  // 33,554,432
    prep_x_kernel<<<total_x / 256, 256, 0, stream>>>(x, xa, total_x);
    prep_w_kernel<<<(2 * 256 * 128) / 256, 256, 0, stream>>>(
        w_ih0, w_hh0, w_ih1, w_hh1, wcat);
    prep_b_kernel<<<2, 256, 0, stream>>>(b_ih0, b_hh0, b_ih1, b_hh1, bsmn);

    lstm_fused_kernel<<<BB / 16, 128, 0, stream>>>(
        xa, wcat, bsmn, fc_w, fc_b, (float*)d_out);
}
